// LearningToSurpriseModel_48911087567578
// MI455X (gfx1250) — matrix operand order repaired and checked by reading, not compiled, not run on hardware
//
#include <hip/hip_runtime.h>
#include <hip/hip_bf16.h>

// GRU LM: B=32, T=1024, V=128, E=512, H=1024
constexpr int B = 32, T = 1024, V = 128, E = 512, H = 1024;
constexpr int G = 3 * H;            // 3072 gate columns (z, r, h)
constexpr int NC = 16;              // hidden columns owned per GRU block
constexpr int NBLK_GRU = H / NC;    // 64 persistent blocks
constexpr int KP = H + 8;           // padded LDS K-stride (bank-conflict avoidance)

// LDS layout of k_gru (dynamic shared, offset 0 = start of allocation)
constexpr unsigned UKB_BYTES = 3u * NC * KP * 2;       // Uk slice, bf16
constexpr unsigned RG_BYTES  = 3u * 32 * NC * 4;       // rg staging, f32
constexpr unsigned XT_OFF    = UKB_BYTES + RG_BYTES;   // xg tile double buffer
constexpr unsigned XT_TILE_B = 32u * 3 * 16 * 2;       // 3072 B per tile

typedef __attribute__((ext_vector_type(16))) __bf16 v16bf;
typedef __attribute__((ext_vector_type(8)))  __bf16 v8bf;
typedef __attribute__((ext_vector_type(8)))  float  v8f;
typedef __attribute__((ext_vector_type(4)))  unsigned u32x4;
typedef __attribute__((ext_vector_type(8)))  int      i32x8;
typedef __attribute__((ext_vector_type(4)))  int      i32x4;

union AFrag { v16bf v; v8bf h[2]; };

// ---------------- TDM: 3D xg gate-slice tile -> LDS ----------------
// Tile: x = 16 contiguous cols, y = 3 gates (stride H), z = 32 batch (stride G).
// LDS receives packed [b][gate][col] bf16 (3 KB).
__device__ __forceinline__ void tdm_load_xg_tile(const __bf16* gsrc, unsigned lds_byte) {
  unsigned long long ga = (unsigned long long)(size_t)gsrc;
  u32x4 g0 = { 1u,                                    // count=1 (valid descriptor)
               lds_byte,                              // lds_addr (bytes)
               (unsigned)(ga & 0xFFFFFFFFu),          // global_addr[31:0]
               (unsigned)((ga >> 32) & 0x1FFFFFFu) | (2u << 30) }; // ga[56:32] | type=2
  i32x8 g1 = { (int)(1u << 16),                       // data_size = 2 bytes
               (int)(16u << 16),                      // tensor_dim0 = 16 (bits 79:48)
               (int)(3u << 16),                       // tensor_dim1 = 3  (bits 111:80)
               (int)(16u << 16),                      // tile_dim0 = 16   (bits 127:112)
               (int)(3u | (32u << 16)),               // tile_dim1 = 3, tile_dim2 = 32
               (int)H,                                // tensor_dim0_stride[31:0] = 1024
               (int)((unsigned)G << 16),              // d0s[47:32]=0 | d1s[15:0]
               (int)((unsigned)G >> 16) };            // tensor_dim1_stride[47:16]
  i32x4 g2 = { 32, 0, 0, 0 };                         // tensor_dim2 = 32
  i32x4 g3 = { 0, 0, 0, 0 };
#if defined(__clang_major__) && __clang_major__ >= 23
  i32x8 gx = { 0, 0, 0, 0, 0, 0, 0, 0 };
  __builtin_amdgcn_tensor_load_to_lds(g0, g1, g2, g3, gx, 0);
#else
  __builtin_amdgcn_tensor_load_to_lds(g0, g1, g2, g3, 0);
#endif
}

// ---------------- conversion / init kernels ----------------

__global__ void k_cvt(const float* __restrict__ src, __bf16* __restrict__ dst, int total) {
  for (int i = blockIdx.x * blockDim.x + threadIdx.x; i < total; i += gridDim.x * blockDim.x)
    dst[i] = (__bf16)src[i];
}

// src row-major [K][N] f32  ->  dst [N][K] bf16 (transposed, for B-side fragments)
__global__ void k_cvtT(const float* __restrict__ src, __bf16* __restrict__ dst, int K, int N) {
  int total = K * N;
  for (int i = blockIdx.x * blockDim.x + threadIdx.x; i < total; i += gridDim.x * blockDim.x) {
    int n = i / K, k = i - n * K;
    dst[i] = (__bf16)src[(size_t)k * N + n];
  }
}

__global__ void k_init(__bf16* __restrict__ h0, int* __restrict__ cnt) {
  int i = blockIdx.x * blockDim.x + threadIdx.x;
  unsigned* p = (unsigned*)h0;
  for (int k = i; k < B * H / 2; k += gridDim.x * blockDim.x) p[k] = 0u;
  if (i == 0) *cnt = 0;
}

// ---------------- phase 1: xg = gather(emb, tokens) @ Wk + b_in  (bf16 out) ----------------

__global__ void k_xg(const int* __restrict__ tokens, const __bf16* __restrict__ embB,
                     const __bf16* __restrict__ WkT, const float* __restrict__ b_in,
                     __bf16* __restrict__ xg) {
  const int lane = threadIdx.x & 31;
  const int w    = threadIdx.x >> 5;
  const int wm   = w >> 1, wn = w & 1;
  const int hi   = lane >> 4;
  const int tileM = blockIdx.y * 64;
  const int tileN = blockIdx.x * 64;

  const int r  = tileM + wm * 16 + (lane & 15);   // r = t*B + b
  const int tt = r >> 5;
  const int bb = r & 31;
  const int tok = tokens[bb * T + tt];
  const __bf16* Arow = embB + (size_t)tok * E;

  const int col0 = tileN + wn * 32 + (lane & 15);
  const int col1 = col0 + 16;
  const __bf16* Bp0 = WkT + (size_t)col0 * E + hi * 16;
  const __bf16* Bp1 = WkT + (size_t)col1 * E + hi * 16;

  v8f a0 = {}, a1 = {};
  for (int kk = 0; kk < E; kk += 32) {
    AFrag av, b0, b1;
    av.h[0] = *(const v8bf*)(Arow + kk + hi * 16);
    av.h[1] = *(const v8bf*)(Arow + kk + hi * 16 + 8);
    b0.h[0] = *(const v8bf*)(Bp0 + kk);
    b0.h[1] = *(const v8bf*)(Bp0 + kk + 8);
    b1.h[0] = *(const v8bf*)(Bp1 + kk);
    b1.h[1] = *(const v8bf*)(Bp1 + kk + 8);
    a0 = __builtin_amdgcn_wmma_f32_16x16x32_bf16(false, av.v, false, b0.v, (short)0, a0, false, false);
    a1 = __builtin_amdgcn_wmma_f32_16x16x32_bf16(false, av.v, false, b1.v, (short)0, a1, false, false);
  }

  const int cm = hi * 8;
  const int cn = lane & 15;
  const int orow = tileM + wm * 16;
  const int cA = tileN + wn * 32 + cn;
#pragma unroll
  for (int i = 0; i < 8; ++i) {
    size_t base = (size_t)(orow + cm + i) * G;
    xg[base + cA]      = (__bf16)(a0[i] + b_in[cA]);
    xg[base + cA + 16] = (__bf16)(a1[i] + b_in[cA + 16]);
  }
}

// ---------------- phase 2: persistent GRU scan ----------------
// 64 blocks x 192 threads (6 waves). Uk slice LDS-resident for all T steps.
// Per step: 32 chained WMMAs per wave; xg gate slice for step t+1 arrives via
// TDM (tensor_load_to_lds) double-buffered in LDS, overlapping the WMMA chain.

__global__ void k_gru(const __bf16* __restrict__ xg, const __bf16* __restrict__ UkT,
                      const float* __restrict__ b_rec, __bf16* __restrict__ h0,
                      __bf16* __restrict__ h1, __bf16* __restrict__ hs,
                      int* __restrict__ cnt) {
  extern __shared__ char smem[];
  __bf16* ukb = (__bf16*)smem;                 // [3][NC][KP] bf16
  float*  rg  = (float*)(smem + UKB_BYTES);    // [3][32][NC] f32
  __bf16* xt  = (__bf16*)(smem + XT_OFF);      // [2][32][3][16] bf16

  const int tid  = threadIdx.x;
  const int lane = tid & 31;
  const int w    = tid >> 5;       // 0..5
  const int g    = w >> 1;         // gate 0..2
  const int mt   = w & 1;          // M tile 0/1
  const int hi   = lane >> 4;
  const int bcol = lane & 15;
  const int colBase = blockIdx.x * NC;

  // Preload this block's Uk slice into LDS (once, reused 1024 steps).
  for (int rr = 0; rr < 3 * NC; ++rr) {
    const __bf16* src = UkT + (size_t)((rr / NC) * H + colBase + (rr % NC)) * H;
    __bf16* dst = ukb + rr * KP;
    const int4* s4 = (const int4*)src;
    int4* d4 = (int4*)dst;
    for (int k = tid; k < H / 8; k += blockDim.x) d4[k] = s4[k];
  }

  // Hoist loop-invariant b_rec slices into registers.
  float br0[3], br1[3], br2[3];
#pragma unroll
  for (int s = 0; s < 3; ++s) {
    int e = tid + s * 192;
    br0[s] = br1[s] = br2[s] = 0.f;
    if (e < 512) {
      int colH = colBase + (e & 15);
      br0[s] = b_rec[colH];
      br1[s] = b_rec[H + colH];
      br2[s] = b_rec[2 * H + colH];
    }
  }

  // Kick off TDM for step 0's xg tile.
  const __bf16* xg_blk = xg + colBase;
  if (w == 0) tdm_load_xg_tile(xg_blk, XT_OFF);
  __syncthreads();

  const int mrow = mt * 16 + (lane & 15);                  // batch row 0..31
  const __bf16* bptr = ukb + (g * NC + bcol) * KP + hi * 16;

  float h_own[3];
#pragma unroll
  for (int s = 0; s < 3; ++s) h_own[s] = 0.f;

  for (int t = 0; t < T; ++t) {
    const __bf16* hA = (t & 1) ? h1 : h0;
    __bf16*       hN = (t & 1) ? h0 : h1;

    // Prefetch next step's xg tile via TDM while this step's WMMAs run.
    if (w == 0 && t + 1 < T)
      tdm_load_xg_tile(xg_blk + (size_t)(t + 1) * B * G, XT_OFF + ((t + 1) & 1) * XT_TILE_B);

    const __bf16* arow = hA + mrow * H;
    v8f acc = {};
    for (int kk = 0; kk < H; kk += 32) {
      AFrag av, bu;
      av.h[0] = *(const v8bf*)(arow + kk + hi * 16);
      av.h[1] = *(const v8bf*)(arow + kk + hi * 16 + 8);
      bu.h[0] = *(const v8bf*)(bptr + kk);
      bu.h[1] = *(const v8bf*)(bptr + kk + 8);
      acc = __builtin_amdgcn_wmma_f32_16x16x32_bf16(false, av.v, false, bu.v, (short)0, acc, false, false);
    }

#pragma unroll
    for (int i = 0; i < 8; ++i)
      rg[(g * 32 + mt * 16 + hi * 8 + i) * NC + bcol] = acc[i];

    // Wave 0 retires the TDM for step t (older op completes in order);
    // the following __syncthreads publishes the LDS tile to all waves.
    if (w == 0) {
      if (t + 1 < T) __builtin_amdgcn_s_wait_tensorcnt(1);
      else           __builtin_amdgcn_s_wait_tensorcnt(0);
    }
    __syncthreads();

    // Elementwise gate math on the 32x16 owned tile (512 elems over 192 threads).
    const __bf16* xts = xt + (t & 1) * (XT_TILE_B / 2);
#pragma unroll
    for (int s = 0; s < 3; ++s) {
      int e = tid + s * 192;
      if (e < 512) {
        int b = e >> 4;
        int c = e & 15;
        int colH = colBase + c;
        float xz = (float)xts[(b * 3 + 0) * 16 + c];
        float xr = (float)xts[(b * 3 + 1) * 16 + c];
        float xh = (float)xts[(b * 3 + 2) * 16 + c];
        float rz = rg[(0 * 32 + b) * NC + c] + br0[s];
        float rr = rg[(1 * 32 + b) * NC + c] + br1[s];
        float rh = rg[(2 * 32 + b) * NC + c] + br2[s];
        float z    = 1.f / (1.f + __expf(-(xz + rz)));
        float rr_g = 1.f / (1.f + __expf(-(xr + rr)));
        float hh = tanhf(xh + rr_g * rh);       // reset_after=True
        float hn = z * h_own[s] + (1.f - z) * hh;
        h_own[s] = hn;
        __bf16 hb = (__bf16)hn;
        hN[b * H + colH] = hb;
        hs[((size_t)t * B + b) * H + colH] = hb;
      }
    }

    // Grid-wide step barrier (monotonic counter; all 64 blocks resident).
    __syncthreads();
    __threadfence();
    if (tid == 0) {
      __hip_atomic_fetch_add(cnt, 1, __ATOMIC_RELEASE, __HIP_MEMORY_SCOPE_AGENT);
      int target = NBLK_GRU * (t + 1);
      while (__hip_atomic_load(cnt, __ATOMIC_ACQUIRE, __HIP_MEMORY_SCOPE_AGENT) < target)
        __builtin_amdgcn_s_sleep(2);
    }
    __syncthreads();
  }
}

// ---------------- phase 3: logits = hs @ Wd + bd ----------------

__global__ void k_logits(const __bf16* __restrict__ hs, const __bf16* __restrict__ WdT,
                         const float* __restrict__ bd, float* __restrict__ out) {
  const int lane = threadIdx.x & 31;
  const int w    = threadIdx.x >> 5;
  const int wm   = w >> 1, wn = w & 1;
  const int hi   = lane >> 4;
  const int tileM = blockIdx.y * 64;
  const int tileN = blockIdx.x * 64;

  const int r = tileM + wm * 16 + (lane & 15);
  const __bf16* Arow = hs + (size_t)r * H;
  const int col0 = tileN + wn * 32 + (lane & 15);
  const int col1 = col0 + 16;
  const __bf16* Bp0 = WdT + (size_t)col0 * H + hi * 16;
  const __bf16* Bp1 = WdT + (size_t)col1 * H + hi * 16;

  v8f a0 = {}, a1 = {};
  for (int kk = 0; kk < H; kk += 32) {
    AFrag av, b0, b1;
    av.h[0] = *(const v8bf*)(Arow + kk + hi * 16);
    av.h[1] = *(const v8bf*)(Arow + kk + hi * 16 + 8);
    b0.h[0] = *(const v8bf*)(Bp0 + kk);
    b0.h[1] = *(const v8bf*)(Bp0 + kk + 8);
    b1.h[0] = *(const v8bf*)(Bp1 + kk);
    b1.h[1] = *(const v8bf*)(Bp1 + kk + 8);
    a0 = __builtin_amdgcn_wmma_f32_16x16x32_bf16(false, av.v, false, b0.v, (short)0, a0, false, false);
    a1 = __builtin_amdgcn_wmma_f32_16x16x32_bf16(false, av.v, false, b1.v, (short)0, a1, false, false);
  }

  const int cm = hi * 8;
  const int cn = lane & 15;
  const int orow = tileM + wm * 16;
  const int cA = tileN + wn * 32 + cn;
#pragma unroll
  for (int i = 0; i < 8; ++i) {
    int row = orow + cm + i;
    int tt = row >> 5, bb = row & 31;
    size_t base = (size_t)bb * T * V + (size_t)tt * V;
    out[base + cA]      = a0[i] + bd[cA];
    out[base + cA + 16] = a1[i] + bd[cA + 16];
  }
}

// ---------------- host launch ----------------

extern "C" void kernel_launch(void* const* d_in, const int* in_sizes, int n_in,
                              void* d_out, int out_size, void* d_ws, size_t ws_size,
                              hipStream_t stream) {
  (void)in_sizes; (void)n_in; (void)out_size; (void)ws_size;
  const int*   tokens = (const int*)d_in[0];
  const float* emb    = (const float*)d_in[1];
  const float* Wk     = (const float*)d_in[2];
  const float* Uk     = (const float*)d_in[3];
  const float* b_in   = (const float*)d_in[4];
  const float* b_rec  = (const float*)d_in[5];
  const float* Wd     = (const float*)d_in[6];
  const float* bd     = (const float*)d_in[7];
  float* out = (float*)d_out;

  char* ws = (char*)d_ws;
  size_t o = 0;
  auto alloc = [&](size_t bytes) { size_t r = o; o += (bytes + 255) & ~(size_t)255; return r; };
  __bf16* embB = (__bf16*)(ws + alloc((size_t)V * E * 2));
  __bf16* WkT  = (__bf16*)(ws + alloc((size_t)G * E * 2));
  __bf16* UkT  = (__bf16*)(ws + alloc((size_t)G * H * 2));
  __bf16* WdT  = (__bf16*)(ws + alloc((size_t)V * H * 2));
  __bf16* xg   = (__bf16*)(ws + alloc((size_t)T * B * G * 2));
  __bf16* h0   = (__bf16*)(ws + alloc((size_t)B * H * 2));
  __bf16* h1   = (__bf16*)(ws + alloc((size_t)B * H * 2));
  __bf16* hsb  = (__bf16*)(ws + alloc((size_t)T * B * H * 2));
  int*    cnt  = (int*)(ws + alloc(256));

  k_cvt <<<dim3(256), dim3(256), 0, stream>>>(emb, embB, V * E);
  k_cvtT<<<dim3(512), dim3(256), 0, stream>>>(Wk, WkT, E, G);
  k_cvtT<<<dim3(512), dim3(256), 0, stream>>>(Uk, UkT, H, G);
  k_cvtT<<<dim3(64),  dim3(256), 0, stream>>>(Wd, WdT, H, V);
  k_init<<<dim3(1),   dim3(256), 0, stream>>>(h0, cnt);

  k_xg<<<dim3(G / 64, (T * B) / 64), dim3(256), 0, stream>>>(tokens, embB, WkT, b_in, xg);

  size_t dynLds = (size_t)UKB_BYTES + RG_BYTES + 2 * XT_TILE_B;
  k_gru<<<dim3(NBLK_GRU), dim3(192), dynLds, stream>>>(xg, UkT, b_rec, h0, h1, hsb, cnt);

  k_logits<<<dim3(V / 64, (T * B) / 64), dim3(256), 0, stream>>>(hsb, WdT, bd, out);
}
